// Correlation_21715354649409
// MI455X (gfx1250) — matrix operand order repaired and checked by reading, not compiled, not run on hardware
//
#include <hip/hip_runtime.h>

// Correlation layer (FlowNet-style), K=1, PAD=20, MAXD=20, S1=1, S2=2.
// out[b, dyi*21+dxi, h, w] = (1/256) * sum_c x1[b,c,h,w] * x2pad[b,c,h+dy,w+dx]
// Per (b,h,dy): Gram tiles G[w,u] = sum_c X1[c,w]*X2[c,u] via v_wmma_f32_16x16x32_f16,
// then scatter the even diagonals dx = u-w in [-20,20].

typedef __attribute__((ext_vector_type(16))) _Float16 v16h;
typedef __attribute__((ext_vector_type(8)))  _Float16 v8h;
typedef __attribute__((ext_vector_type(8)))  float    v8f;

constexpr int Bn   = 8;
constexpr int Cn   = 256;
constexpr int Hn   = 64;
constexpr int Wn   = 96;
constexpr int GRID = 21;      // displacements per axis, dy/dx in {-20..20 step 2}
constexpr int ULEN = 144;     // padded u range: u in [-20, 124)
constexpr int LSTR = 264;     // LDS channel-stride (f16 units), padded for bank spread
constexpr int NTHREADS = 192; // 6 waves; wave i owns w-tile [16i, 16i+16)

__global__ __launch_bounds__(NTHREADS)
void corr_wmma_kernel(const float* __restrict__ x1,
                      const float* __restrict__ x2,
                      float* __restrict__ out) {
  extern __shared__ _Float16 smem[];
  _Float16* x1s = smem;                 // [Wn][LSTR]  : x1 row h (w-major, c-minor)
  _Float16* x2s = smem + Wn * LSTR;     // [ULEN][LSTR]: x2 row h+dy, zero-padded in u

  const int tid  = threadIdx.x;
  const int lane = tid & 31;
  const int wave = tid >> 5;            // 0..5
  const int bh   = blockIdx.x;          // b*Hn + h
  const int b    = bh / Hn;
  const int h    = bh % Hn;

  const int Nn = lane & 15;             // N column within tile
  const int hi = lane >> 4;             // lane-half selects K/M split
  const int w0 = wave * 16;

  // ---- stage x1[b,:,h,:] -> LDS [w][c], float4 loads, f32->f16 ----
  {
    const float* src = x1 + ((size_t)b * Cn) * Hn * Wn + (size_t)h * Wn;
    for (int i = tid; i < Cn * (Wn / 4); i += NTHREADS) {   // 32 uniform iters
      int c = i / (Wn / 4);
      int w = (i % (Wn / 4)) * 4;
      const float4 v = *(const float4*)(src + (size_t)c * Hn * Wn + w);
      int base = w * LSTR + c;
      x1s[base           ] = (_Float16)v.x;
      x1s[base +     LSTR] = (_Float16)v.y;
      x1s[base + 2 * LSTR] = (_Float16)v.z;
      x1s[base + 3 * LSTR] = (_Float16)v.w;
    }
  }

  // ---- one-time zero-fill of x2s pad rows (u_lds < 20 or >= 116), b128 stores ----
  {
    const v8h z = {};
    for (int i = tid; i < 48 * (Cn / 8); i += NTHREADS) {   // 8 uniform iters
      int row = i / (Cn / 8);
      int ul  = (row < 20) ? row : (row + 96);              // 0..19 and 116..143
      int cc  = (i % (Cn / 8)) * 8;
      *(v8h*)&x2s[ul * LSTR + cc] = z;
    }
  }

  __syncthreads();  // x1s complete for A-fragment hoist

  // ---- hoist all A fragments into registers (invariant across dy) ----
  // A (16x32 f16, M=w, K=c): lanes 0-15: K=c0+{0..7},{16..23}; lanes 16-31: +8
  v16h afrag[8];
  #pragma unroll
  for (int ks = 0; ks < Cn / 32; ++ks) {
    const _Float16* ap = &x1s[(w0 + Nn) * LSTR + ks * 32 + hi * 8];
    v8h alo = *(const v8h*)ap;
    v8h ahi = *(const v8h*)(ap + 16);
    afrag[ks] = __builtin_shufflevector(alo, ahi,
                0,1,2,3,4,5,6,7,8,9,10,11,12,13,14,15);
  }

  for (int dyi = 0; dyi < GRID; ++dyi) {
    const int h2 = h + (dyi * 2 - 20);
    __syncthreads();  // previous iteration's x2s readers done before overwrite

    if (h2 < 0 || h2 >= Hn) {
      // fully zero-padded shifted slice -> this dy's 21x96 output rows are 0
      for (int i = tid; i < GRID * Wn; i += NTHREADS) {
        int dxi = i / Wn;
        int w   = i % Wn;
        out[(((size_t)(b * GRID * GRID + dyi * GRID + dxi)) * Hn + h) * Wn + w] = 0.0f;
      }
      continue;  // block-uniform branch
    }

    // ---- stage valid region of x2[b,:,h2,:] -> x2s[u+20][c], float4 loads ----
    {
      const float* src = x2 + ((size_t)b * Cn) * Hn * Wn + (size_t)h2 * Wn;
      for (int i = tid; i < Cn * (Wn / 4); i += NTHREADS) { // 32 uniform iters
        int c = i / (Wn / 4);
        int u = (i % (Wn / 4)) * 4;
        const float4 v = *(const float4*)(src + (size_t)c * Hn * Wn + u);
        int base = (u + 20) * LSTR + c;
        x2s[base           ] = (_Float16)v.x;
        x2s[base +     LSTR] = (_Float16)v.y;
        x2s[base + 2 * LSTR] = (_Float16)v.z;
        x2s[base + 3 * LSTR] = (_Float16)v.w;
      }
    }
    __syncthreads();

    // ---- prefetch next dy's x2 row while we compute (global_prefetch_b8) ----
    {
      const int h2n = h2 + 2;
      if (h2n < Hn) {
        const float* nsrc = x2 + ((size_t)b * Cn) * Hn * Wn + (size_t)h2n * Wn;
        for (int i = tid; i < Cn * 3; i += NTHREADS) {      // 4 uniform iters
          int c    = i / 3;
          int part = i % 3;                                  // 3 x 128B lines per row
          __builtin_prefetch(nsrc + (size_t)c * Hn * Wn + part * 32, 0, 3);
        }
      }
    }

    // ---- Gram tiles: G[w0..w0+15, w0+16j .. +15] for j=0..3 ----
    v8f acc[4] = {v8f{}, v8f{}, v8f{}, v8f{}};

    #pragma unroll
    for (int ks = 0; ks < Cn / 32; ++ks) {
      const int c0 = ks * 32;
      #pragma unroll
      for (int j = 0; j < 4; ++j) {
        // B (32x16 f16, K=c, N=u): lanes 0-15: K=c0+{0..7},{16..23}; lanes 16-31: +8
        const _Float16* bp = &x2s[(w0 + 16 * j + Nn) * LSTR + c0 + hi * 8];
        v8h blo = *(const v8h*)bp;
        v8h bhi = *(const v8h*)(bp + 16);
        v16h bmat = __builtin_shufflevector(blo, bhi,
                    0,1,2,3,4,5,6,7,8,9,10,11,12,13,14,15);

        acc[j] = __builtin_amdgcn_wmma_f32_16x16x32_f16(
            /*neg_a=*/false, afrag[ks], /*neg_b=*/false, bmat,
            /*c_mod=*/(short)0, acc[j], /*reuse_a=*/false, /*reuse_b=*/false);
      }
    }

    // ---- harvest even diagonals: dx = 16j - 20 + N - M, |dx| <= 20, dx even ----
    // D layout: VGPR r holds row M = r + 8*hi at column N = lane & 15.
    const float scale = 1.0f / 256.0f;
    #pragma unroll
    for (int j = 0; j < 4; ++j) {
      #pragma unroll
      for (int r = 0; r < 8; ++r) {
        int M  = r + 8 * hi;
        int dx = 16 * j - 20 + Nn - M;
        if (((dx & 1) == 0) && dx >= -20 && dx <= 20) {
          int dxi = (dx + 20) >> 1;
          int w   = w0 + M;
          out[(((size_t)(b * GRID * GRID + dyi * GRID + dxi)) * Hn + h) * Wn + w]
              = acc[j][r] * scale;
        }
      }
    }
  }
}

extern "C" void kernel_launch(void* const* d_in, const int* in_sizes, int n_in,
                              void* d_out, int out_size, void* d_ws, size_t ws_size,
                              hipStream_t stream) {
  const float* x1 = (const float*)d_in[0];
  const float* x2 = (const float*)d_in[1];
  float* out = (float*)d_out;

  const size_t smem_bytes = (size_t)(Wn * LSTR + ULEN * LSTR) * sizeof(_Float16); // ~124 KB
  // Dynamic LDS > 64 KB: raise the cap (host-side, idempotent, graph-capture safe).
  (void)hipFuncSetAttribute(reinterpret_cast<const void*>(corr_wmma_kernel),
                            hipFuncAttributeMaxDynamicSharedMemorySize,
                            (int)smem_bytes);

  corr_wmma_kernel<<<Bn * Hn, NTHREADS, smem_bytes, stream>>>(x1, x2, out);
}
